// AttNHPTransformerLayer_91225105367373
// MI455X (gfx1250) — matrix operand order repaired and checked by reading, not compiled, not run on hardware
//
#include <hip/hip_runtime.h>
#include <hip/hip_bf16.h>

// ---------------------------------------------------------------------------
// Transformer encoder layer (B=4, L=1024, D=512, H=8, DV=64, F=2048) on
// MI455X (gfx1250). All GEMMs use v_wmma_f32_16x16x32_f16 (f16 in, f32 acc).
// GEMM activation tiles staged block-wide via TENSOR_LOAD_TO_LDS (TDM);
// attention Q tiles staged with global_load_async_to_lds_b128 (ASYNCcnt).
// Workspace usage ~130MB (assumed available in d_ws).
// ---------------------------------------------------------------------------

typedef _Float16 h16 __attribute__((ext_vector_type(16)));
typedef _Float16 h8  __attribute__((ext_vector_type(8)));
typedef float    f8  __attribute__((ext_vector_type(8)));
typedef unsigned int u32x4 __attribute__((ext_vector_type(4)));
typedef int          i32x4 __attribute__((ext_vector_type(4)));
typedef int          i32x8 __attribute__((ext_vector_type(8)));

#define BATCH 4
#define SEQL  1024
#define DMODEL 512
#define NHEAD 8
#define DVAL  64
#define DFF   2048
#define PROJ  8704   // 2*H*D + H*DV
#define NTOK  4096   // BATCH*SEQL
#define NINF_V (-1000000.0f)
#define QSCALE 0.044194173824159216f   // 1/sqrt(512)

// ---- WMMA fragment helpers (layouts per CDNA5 ISA 7.12.2, wave32) ---------

// A-matrix 16x32 f16: lanes 0-15 rows M=0-15 (K 0..7 then 16..23),
// lanes 16-31 same rows (K 8..15 then 24..31).
__device__ inline h16 load_a_frag(const _Float16* A, int lda, int row0, int k0, int lane) {
    int laneLo = lane & 15, laneHi = lane >> 4;
    const _Float16* p = A + (size_t)(row0 + laneLo) * lda + k0 + 8 * laneHi;
    h8 lo = *(const h8*)p;          // K = k0 + 8*hi   .. +8
    h8 hi = *(const h8*)(p + 16);   // K = k0+16+8*hi  .. +8
    return __builtin_shufflevector(lo, hi, 0,1,2,3,4,5,6,7,8,9,10,11,12,13,14,15);
}

// B-matrix 32x16 f16, B = W^T with W row-major (N x K): column n of B is row n
// of W; lanes 0-15 hold K 0..7 then 16..23, lanes 16-31 K 8..15 then 24..31.
__device__ inline h16 load_b_frag(const _Float16* W, int ldw, int n0, int k0, int lane) {
    int laneLo = lane & 15, laneHi = lane >> 4;
    const _Float16* p = W + (size_t)(n0 + laneLo) * ldw + k0 + 8 * laneHi;
    h8 lo = *(const h8*)p;
    h8 hi = *(const h8*)(p + 16);
    return __builtin_shufflevector(lo, hi, 0,1,2,3,4,5,6,7,8,9,10,11,12,13,14,15);
}

__device__ inline f8 wmma_f16(h16 a, h16 b, f8 c) {
    return __builtin_amdgcn_wmma_f32_16x16x32_f16(
        /*neg_a=*/false, a, /*neg_b=*/false, b,
        /*c_mod=*/(short)0, c, /*reuse_a=*/false, /*reuse_b=*/false);
}

// Row reductions across the 16 lanes that share a C-matrix row.
__device__ inline float rowmax16(float v) {
    #pragma unroll
    for (int m = 8; m >= 1; m >>= 1) v = fmaxf(v, __shfl_xor(v, m, 16));
    return v;
}
__device__ inline float rowsum16(float v) {
    #pragma unroll
    for (int m = 8; m >= 1; m >>= 1) v += __shfl_xor(v, m, 16);
    return v;
}

// ---- CDNA5 async global -> LDS (GLOBAL_LOAD_ASYNC_TO_LDS_B128, ASYNCcnt) --

__device__ inline void async_copy_b128(unsigned ldsoff, unsigned long long gaddr) {
    asm volatile("global_load_async_to_lds_b128 %0, %1, off"
                 :: "v"(ldsoff), "v"(gaddr) : "memory");
}
__device__ inline void wait_asynccnt0() {
    asm volatile("s_wait_asynccnt 0" ::: "memory");
}

// ---- CDNA5 Tensor Data Mover: 2D tile of f16 global -> LDS ----------------
// D# per ISA sec 8.3/8.4: group0 = {count=1, lds_addr, global_addr, type=2},
// group1 = {data_size=2B, tensor_dim0/1, tile_dim0/1, tensor_dim0_stride}.
// Issued by one wave; completion via TENSORcnt.

__device__ inline void tdm_load_tile_f16(const _Float16* lds_dst, const _Float16* gsrc,
                                         unsigned rows, unsigned cols,
                                         unsigned row_stride_elems) {
    unsigned lds_off = (unsigned)(unsigned long long)lds_dst;  // low 32b = LDS byte addr
    unsigned long long ga = (unsigned long long)gsrc;

    u32x4 g0;
    g0.x = 1u;                                                 // count=1 (user D#)
    g0.y = (unsigned)__builtin_amdgcn_readfirstlane((int)lds_off);
    g0.z = (unsigned)__builtin_amdgcn_readfirstlane((int)(unsigned)ga);
    g0.w = ((unsigned)__builtin_amdgcn_readfirstlane((int)(unsigned)(ga >> 32)) & 0x01FFFFFFu)
           | (2u << 30);                                       // type=2 ("image")

    i32x8 g1;
    g1[0] = (int)(1u << 16);                                   // data_size=1 -> 2 bytes
    g1[1] = (int)((cols & 0xFFFFu) << 16);                     // tensor_dim0[15:0]
    g1[2] = (int)(((cols >> 16) & 0xFFFFu) | ((rows & 0xFFFFu) << 16)); // td0 hi | td1 lo
    g1[3] = (int)(((rows >> 16) & 0xFFFFu) | ((cols & 0xFFFFu) << 16)); // td1 hi | tile_dim0
    g1[4] = (int)(rows & 0xFFFFu);                             // tile_dim1 (tile_dim2=0)
    g1[5] = (int)row_stride_elems;                             // tensor_dim0_stride lo32
    g1[6] = 0;                                                 // stride hi / dim1_stride
    g1[7] = 0;

    i32x4 z4 = {0, 0, 0, 0};
#if defined(__clang_major__) && __clang_major__ >= 23
    i32x8 z8 = {0, 0, 0, 0, 0, 0, 0, 0};
    __builtin_amdgcn_tensor_load_to_lds(g0, g1, z4, z4, z8, 0);
#else
    __builtin_amdgcn_tensor_load_to_lds(g0, g1, z4, z4, 0);
#endif
}

__device__ inline void wait_tensorcnt0() {
    __builtin_amdgcn_s_wait_tensorcnt(0);
}

// ---- fp32 -> f16 conversion ----------------------------------------------

__global__ void cvt_f32_f16_kernel(const float* __restrict__ in,
                                   _Float16* __restrict__ out, int n) {
    int i = blockIdx.x * blockDim.x + threadIdx.x;
    if (i < n) out[i] = (_Float16)in[i];
}

// ---- Shared GEMM body: block computes 32 rows x 512 cols ------------------
// 8 waves share a 32xKCHUNK X tile staged in LDS by the TDM; each wave owns a
// 32x64 output tile (2 A-frags reuse each B-frag). KD = full contraction dim.

template <int KD>
__device__ inline void gemm_body(const _Float16* __restrict__ X,
                                 const _Float16* __restrict__ W,
                                 _Float16* xs,          // LDS, 32*512 halves
                                 int row0, int n0, int lane, int wave,
                                 f8 acc[8]) {
    for (int kc = 0; kc < KD; kc += 512) {
        if (kc) __syncthreads();                       // protect LDS reuse (WAR)
        if (wave == 0) {
            tdm_load_tile_f16(xs, X + (size_t)row0 * KD + kc, 32, 512, KD);
            wait_tensorcnt0();
        }
        __syncthreads();
        for (int k0 = 0; k0 < 512; k0 += 32) {
            h16 a0 = load_a_frag(xs, 512, 0,  k0, lane);
            h16 a1 = load_a_frag(xs, 512, 16, k0, lane);
            if (k0 + 32 < 512)
                __builtin_prefetch(W + (size_t)(n0 + (lane & 15)) * KD + kc + k0 + 32, 0, 1);
            #pragma unroll
            for (int j = 0; j < 4; ++j) {
                h16 b = load_b_frag(W, KD, n0 + 16 * j, kc + k0, lane);
                acc[j]     = wmma_f16(a0, b, acc[j]);
                acc[4 + j] = wmma_f16(a1, b, acc[4 + j]);
            }
        }
    }
}

// ---- QKV projection GEMM: (4096 x 512) * (512 x 8704) + bias --------------

__global__ __launch_bounds__(256)
void gemm_qkv_kernel(const _Float16* __restrict__ X, const _Float16* __restrict__ W,
                     const float* __restrict__ bias,
                     _Float16* __restrict__ q, _Float16* __restrict__ k,
                     _Float16* __restrict__ vT) {
    __shared__ __attribute__((aligned(128))) _Float16 xs[32 * 512];
    const int CB = (PROJ / 64) / 8;                 // 17 column blocks
    int lane = threadIdx.x & 31, wave = threadIdx.x >> 5;
    int row0 = (blockIdx.x / CB) * 32;
    int n0   = ((blockIdx.x % CB) * 8 + wave) * 64;

    f8 acc[8] = {};
    gemm_body<DMODEL>(X, W, xs, row0, n0, lane, wave, acc);

    int laneLo = lane & 15, laneHi = lane >> 4;
    #pragma unroll
    for (int mi = 0; mi < 2; ++mi) {
        #pragma unroll
        for (int j = 0; j < 4; ++j) {
            int col = n0 + 16 * j + laneLo;
            float bv = bias[col];
            #pragma unroll
            for (int r = 0; r < 8; ++r) {
                int row = row0 + 16 * mi + r + 8 * laneHi;  // token index 0..4095
                int bb = row >> 10, l = row & 1023;
                float v = acc[4 * mi + j][r] + bv;
                if (col < NHEAD * DMODEL) {             // Q
                    int h = col >> 9, d = col & 511;
                    q[(((size_t)(bb * NHEAD + h)) * SEQL + l) * DMODEL + d] =
                        (_Float16)(v * QSCALE);
                } else if (col < 2 * NHEAD * DMODEL) {  // K
                    int c = col - NHEAD * DMODEL;
                    int h = c >> 9, d = c & 511;
                    k[(((size_t)(bb * NHEAD + h)) * SEQL + l) * DMODEL + d] = (_Float16)v;
                } else {                                // V, stored transposed DVxL
                    int c = col - 2 * NHEAD * DMODEL;
                    int h = c >> 6, d = c & 63;
                    vT[(((size_t)(bb * NHEAD + h)) * DVAL + d) * SEQL + l] = (_Float16)v;
                }
            }
        }
    }
}

// ---- Flash attention: per (b,h), 16 query rows per wave -------------------

__global__ __launch_bounds__(64)
void attn_kernel(const _Float16* __restrict__ qh, const _Float16* __restrict__ kh,
                 const _Float16* __restrict__ vT,
                 const unsigned char* __restrict__ kmask,   // (B,L) bool
                 const unsigned char* __restrict__ amask,   // (L,L) bool
                 float* __restrict__ sa) {
    __shared__ __attribute__((aligned(32))) _Float16 ldsQ[2 * 16 * DMODEL];
    __shared__ __attribute__((aligned(32))) _Float16 ldsP[2 * 16 * 32];

    int lane = threadIdx.x & 31, wave = threadIdx.x >> 5;
    int tile = blockIdx.x * 2 + wave;        // 2048 row-tiles total
    int bh = tile >> 6;                      // (b*H + h)
    int rowbase = (tile & 63) << 4;
    int b = bh >> 3, h = bh & 7;

    const _Float16* Q = qh + (size_t)bh * SEQL * DMODEL;
    const _Float16* K = kh + (size_t)bh * SEQL * DMODEL;
    const _Float16* V = vT + (size_t)bh * DVAL * SEQL;
    _Float16* q_s = ldsQ + wave * 16 * DMODEL;
    _Float16* p_s = ldsP + wave * 16 * 32;

    // stage this wave's 16 Q rows (contiguous 16KB region) into LDS with the
    // async-to-LDS path (ASYNCcnt), 16B per lane per issue.
    {
        unsigned qs_base = (unsigned)(unsigned long long)q_s;     // LDS byte offset
        unsigned long long gbase =
            (unsigned long long)(Q + (size_t)rowbase * DMODEL);
        for (int i = lane; i < 16 * DMODEL * 2 / 16; i += 32)     // 1024 chunks
            async_copy_b128(qs_base + i * 16, gbase + (unsigned long long)i * 16);
        wait_asynccnt0();
    }
    __syncthreads();

    int laneLo = lane & 15, laneHi = lane >> 4;
    const unsigned char* km = kmask + b * SEQL;

    f8 o0 = {}, o1 = {}, o2 = {}, o3 = {};
    float mrow[8], lrow[8];
    #pragma unroll
    for (int r = 0; r < 8; ++r) { mrow[r] = -3e38f; lrow[r] = 0.0f; }

    for (int m0 = 0; m0 < SEQL; m0 += 32) {
        // S tile 16x32 = Q(16x512) * K^T chunk
        f8 c0 = {}, c1 = {};
        for (int k0 = 0; k0 < DMODEL; k0 += 32) {
            h16 a  = load_a_frag(q_s, DMODEL, 0, k0, lane);
            h16 b0 = load_b_frag(K, DMODEL, m0,      k0, lane);
            h16 b1 = load_b_frag(K, DMODEL, m0 + 16, k0, lane);
            c0 = wmma_f16(a, b0, c0);
            c1 = wmma_f16(a, b1, c1);
        }

        int col0 = m0 + laneLo, col1 = col0 + 16;
        bool km0 = km[col0] != 0, km1 = km[col1] != 0;

        float p0[8], p1[8], alpha[8];
        #pragma unroll
        for (int r = 0; r < 8; ++r) {
            int l = rowbase + r + 8 * laneHi;
            float s0 = c0[r], s1 = c1[r];
            if (km0 || amask[(size_t)l * SEQL + col0]) s0 = NINF_V;
            if (km1 || amask[(size_t)l * SEQL + col1]) s1 = NINF_V;
            float cm   = rowmax16(fmaxf(s0, s1));
            float mnew = fmaxf(mrow[r], cm);
            float al   = __expf(mrow[r] - mnew);
            float e0 = __expf(s0 - mnew), e1 = __expf(s1 - mnew);
            lrow[r] = lrow[r] * al + rowsum16(e0 + e1);
            mrow[r] = mnew;
            alpha[r] = al;
            p0[r] = e0; p1[r] = e1;
        }
        #pragma unroll
        for (int r = 0; r < 8; ++r) {
            o0[r] *= alpha[r]; o1[r] *= alpha[r];
            o2[r] *= alpha[r]; o3[r] *= alpha[r];
        }

        // bounce P (C-layout) through LDS to reshape into an A-fragment
        #pragma unroll
        for (int r = 0; r < 8; ++r) {
            p_s[(r + 8 * laneHi) * 32 + laneLo]      = (_Float16)p0[r];
            p_s[(r + 8 * laneHi) * 32 + laneLo + 16] = (_Float16)p1[r];
        }
        __syncthreads();

        h16 aP = load_a_frag(p_s, 32, 0, 0, lane);
        h16 v0 = load_b_frag(V, SEQL, 0,  m0, lane);
        h16 v1 = load_b_frag(V, SEQL, 16, m0, lane);
        h16 v2 = load_b_frag(V, SEQL, 32, m0, lane);
        h16 v3 = load_b_frag(V, SEQL, 48, m0, lane);
        o0 = wmma_f16(aP, v0, o0);
        o1 = wmma_f16(aP, v1, o1);
        o2 = wmma_f16(aP, v2, o2);
        o3 = wmma_f16(aP, v3, o3);
        __syncthreads();
    }

    // normalize and write sa in (B, L, H*DV) layout; zero fully-masked rows
    #pragma unroll
    for (int r = 0; r < 8; ++r) {
        bool valid = (mrow[r] > -1e5f) && (lrow[r] > 0.0f);
        float inv = valid ? 1.0f / lrow[r] : 0.0f;
        int l = rowbase + r + 8 * laneHi;
        float* dst = sa + ((size_t)(b * SEQL + l)) * DMODEL + h * DVAL + laneLo;
        dst[0]  = o0[r] * inv;
        dst[16] = o1[r] * inv;
        dst[32] = o2[r] * inv;
        dst[48] = o3[r] * inv;
    }
}

// ---- residual + LayerNorm (row = one (b,l) token of 512 elems) ------------

__global__ __launch_bounds__(256)
void ln_kernel(const float* __restrict__ a, const float* __restrict__ res,
               const float* __restrict__ g, const float* __restrict__ bb,
               float* __restrict__ out, _Float16* __restrict__ outh) {
    __shared__ float s[256];
    int row = blockIdx.x, t = threadIdx.x;
    size_t base = (size_t)row * DMODEL;
    float x0 = a[base + t] + res[base + t];
    float x1 = a[base + t + 256] + res[base + t + 256];

    s[t] = x0 + x1;
    __syncthreads();
    for (int off = 128; off > 0; off >>= 1) {
        if (t < off) s[t] += s[t + off];
        __syncthreads();
    }
    float mu = s[0] * (1.0f / DMODEL);
    __syncthreads();

    float d0 = x0 - mu, d1 = x1 - mu;
    s[t] = d0 * d0 + d1 * d1;
    __syncthreads();
    for (int off = 128; off > 0; off >>= 1) {
        if (t < off) s[t] += s[t + off];
        __syncthreads();
    }
    float inv = rsqrtf(s[0] * (1.0f / DMODEL) + 1e-5f);

    float y0 = d0 * inv * g[t] + bb[t];
    float y1 = d1 * inv * g[t + 256] + bb[t + 256];
    out[base + t] = y0;
    out[base + t + 256] = y1;
    if (outh) {
        outh[base + t] = (_Float16)y0;
        outh[base + t + 256] = (_Float16)y1;
    }
}

// ---- FFN GEMMs (32x64 per wave, TDM-staged X tile) ------------------------

__global__ __launch_bounds__(256)
void gemm_ff1_kernel(const _Float16* __restrict__ X, const _Float16* __restrict__ W,
                     const float* __restrict__ bias, _Float16* __restrict__ out) {
    __shared__ __attribute__((aligned(128))) _Float16 xs[32 * 512];
    const int CB = (DFF / 64) / 8;                  // 4 column blocks
    int lane = threadIdx.x & 31, wave = threadIdx.x >> 5;
    int row0 = (blockIdx.x / CB) * 32;
    int n0   = ((blockIdx.x % CB) * 8 + wave) * 64;

    f8 acc[8] = {};
    gemm_body<DMODEL>(X, W, xs, row0, n0, lane, wave, acc);

    int laneLo = lane & 15, laneHi = lane >> 4;
    #pragma unroll
    for (int mi = 0; mi < 2; ++mi) {
        #pragma unroll
        for (int j = 0; j < 4; ++j) {
            int col = n0 + 16 * j + laneLo;
            float bv = bias[col];
            #pragma unroll
            for (int r = 0; r < 8; ++r) {
                int row = row0 + 16 * mi + r + 8 * laneHi;
                float v = fmaxf(acc[4 * mi + j][r] + bv, 0.0f);  // ReLU
                out[(size_t)row * DFF + col] = (_Float16)v;
            }
        }
    }
}

__global__ __launch_bounds__(256)
void gemm_ff2_kernel(const _Float16* __restrict__ X, const _Float16* __restrict__ W,
                     const float* __restrict__ bias, float* __restrict__ out) {
    __shared__ __attribute__((aligned(128))) _Float16 xs[32 * 512];
    // 8 column groups of 64 -> exactly one column block of 8 waves per block
    int lane = threadIdx.x & 31, wave = threadIdx.x >> 5;
    int row0 = blockIdx.x * 32;
    int n0   = wave * 64;

    f8 acc[8] = {};
    gemm_body<DFF>(X, W, xs, row0, n0, lane, wave, acc);

    int laneLo = lane & 15, laneHi = lane >> 4;
    #pragma unroll
    for (int mi = 0; mi < 2; ++mi) {
        #pragma unroll
        for (int j = 0; j < 4; ++j) {
            int col = n0 + 16 * j + laneLo;
            float bv = bias[col];
            #pragma unroll
            for (int r = 0; r < 8; ++r) {
                int row = row0 + 16 * mi + r + 8 * laneHi;
                out[(size_t)row * DMODEL + col] = acc[4 * mi + j][r] + bv;
            }
        }
    }
}

// ---------------------------------------------------------------------------

extern "C" void kernel_launch(void* const* d_in, const int* in_sizes, int n_in,
                              void* d_out, int out_size, void* d_ws, size_t ws_size,
                              hipStream_t stream) {
    (void)in_sizes; (void)n_in; (void)out_size; (void)ws_size;

    const float*         src   = (const float*)d_in[0];
    const unsigned char* mask  = (const unsigned char*)d_in[1];   // jnp bool = 1B
    const unsigned char* amask = (const unsigned char*)d_in[2];
    const float* wqkv = (const float*)d_in[3];
    const float* bqkv = (const float*)d_in[4];
    const float* g1   = (const float*)d_in[5];
    const float* b1   = (const float*)d_in[6];
    const float* w1   = (const float*)d_in[7];
    const float* bf1  = (const float*)d_in[8];
    const float* w2   = (const float*)d_in[9];
    const float* bf2  = (const float*)d_in[10];
    const float* g2   = (const float*)d_in[11];
    const float* b2   = (const float*)d_in[12];
    float* out = (float*)d_out;

    char* p = (char*)d_ws;
    auto carve = [&](size_t bytes) -> void* {
        void* r = (void*)p;
        p += (bytes + 255) & ~(size_t)255;
        return r;
    };
    _Float16* Xh  = (_Float16*)carve((size_t)NTOK * DMODEL * 2);
    _Float16* Wph = (_Float16*)carve((size_t)PROJ * DMODEL * 2);
    _Float16* W1h = (_Float16*)carve((size_t)DFF * DMODEL * 2);
    _Float16* W2h = (_Float16*)carve((size_t)DMODEL * DFF * 2);
    _Float16* qh  = (_Float16*)carve((size_t)BATCH * NHEAD * SEQL * DMODEL * 2);
    _Float16* kh  = (_Float16*)carve((size_t)BATCH * NHEAD * SEQL * DMODEL * 2);
    _Float16* vTh = (_Float16*)carve((size_t)BATCH * NHEAD * DVAL * SEQL * 2);
    float*    sa  = (float*)carve((size_t)NTOK * DMODEL * 4);
    float*    x1  = (float*)carve((size_t)NTOK * DMODEL * 4);
    _Float16* x1h = (_Float16*)carve((size_t)NTOK * DMODEL * 2);
    _Float16* ffh = (_Float16*)carve((size_t)NTOK * DFF * 2);
    float*    f2o = (float*)carve((size_t)NTOK * DMODEL * 4);

    // fp32 -> f16 conversions
    int n;
    n = NTOK * DMODEL;
    cvt_f32_f16_kernel<<<(n + 255) / 256, 256, 0, stream>>>(src, Xh, n);
    n = PROJ * DMODEL;
    cvt_f32_f16_kernel<<<(n + 255) / 256, 256, 0, stream>>>(wqkv, Wph, n);
    n = DFF * DMODEL;
    cvt_f32_f16_kernel<<<(n + 255) / 256, 256, 0, stream>>>(w1, W1h, n);
    n = DMODEL * DFF;
    cvt_f32_f16_kernel<<<(n + 255) / 256, 256, 0, stream>>>(w2, W2h, n);

    // QKV projection: 128 row groups x 17 column blocks
    gemm_qkv_kernel<<<128 * ((PROJ / 64) / 8), 256, 0, stream>>>(
        Xh, Wph, bqkv, qh, kh, vTh);

    // attention: 2048 row-tiles, 2 waves/block
    attn_kernel<<<1024, 64, 0, stream>>>(qh, kh, vTh, mask, amask, sa);

    // x1 = LN(src + sa)
    ln_kernel<<<NTOK, 256, 0, stream>>>(sa, src, g1, b1, x1, x1h);

    // FFN
    gemm_ff1_kernel<<<128 * ((DFF / 64) / 8), 256, 0, stream>>>(x1h, W1h, bf1, ffh);
    gemm_ff2_kernel<<<128, 256, 0, stream>>>(ffh, W2h, bf2, f2o);

    // out = LN(x1 + ff)
    ln_kernel<<<NTOK, 256, 0, stream>>>(f2o, x1, g2, b2, out, nullptr);
}
